// Net_49520972923534
// MI455X (gfx1250) — matrix operand order repaired and checked by reading, not compiled, not run on hardware
//
#include <hip/hip_runtime.h>
#include <hip/hip_bf16.h>
#include <math.h>

// ---------------- problem constants (match reference) ----------------
#define PN   100000
#define PE   1600000
#define F_IN 32
#define PD   128
#define PL   5
#define PB   3000
#define PV   64
#define PT   128
#define PEMB 64
#define PH   640          // L*D
#define PH4  2560         // 4*H
#define PKC  (PEMB + PH)  // 704 = concatenated LSTM K
#define BN_EPS 1e-5f

// ---------------- WMMA types ----------------
typedef __bf16 v16bf __attribute__((ext_vector_type(16)));
typedef float  v8f   __attribute__((ext_vector_type(8)));

union FragU {
    uint4 q[2];
    unsigned short u[16];
    v16bf v;
};

__device__ __forceinline__ unsigned short f2bf(float f) {
    unsigned int u = __float_as_uint(f);
    unsigned int r = (u + 0x7FFFu + ((u >> 16) & 1u)) >> 16;   // RNE
    return (unsigned short)r;
}
__device__ __forceinline__ float sigm(float x) { return 1.0f / (1.0f + __expf(-x)); }

// permute K index into the 16-bit A-fragment storage order:
// positions [0..15] hold K {0..7, 16..23} (lane-half 0), [16..31] hold K {8..15, 24..31}
__device__ __forceinline__ int kperm(int k) {
    return k + (((k >> 3) == 1) ? 8 : ((k >> 3) == 2) ? -8 : 0);
}

// ---------------- utility kernels ----------------
__global__ void k_zero(float* p, size_t n) {
    size_t i = (size_t)blockIdx.x * blockDim.x + threadIdx.x;
    if (i < n) p[i] = 0.0f;
}

__global__ void k_addvec(float* dst, const float* a, const float* b, int n) {
    int i = blockIdx.x * blockDim.x + threadIdx.x;
    if (i < n) dst[i] = a[i] + b[i];
}

// fp32 -> bf16 convert with dst row-stride/offset (supports concatenated weights).
// transpose==0: dst[r*dstStride + dstOff + c] = src[r][c]
// transpose==1: dst[c*dstStride + dstOff + r] = src[r][c]
__global__ void k_conv_bf16(const float* src, unsigned short* dst, int R, int C,
                            int transpose, int dstStride, int dstOff) {
    int i = blockIdx.x * blockDim.x + threadIdx.x;
    if (i >= R * C) return;
    int r = i / C, c = i % C;
    unsigned short v = f2bf(src[i]);
    if (transpose) dst[(size_t)c * dstStride + dstOff + r] = v;
    else           dst[(size_t)r * dstStride + dstOff + c] = v;
}

// agg[dst[e]] += h[src[e]] ; one thread per edge x 4 features (float4 read, 4 atomics)
__global__ void k_edge_agg(const float* __restrict__ h, const int* __restrict__ src,
                           const int* __restrict__ dst, float* __restrict__ agg,
                           int E, int Dc) {
    int qn = Dc >> 2;
    int i = blockIdx.x * blockDim.x + threadIdx.x;
    if (i >= E * qn) return;
    int e = i / qn, q = (i - e * qn) << 2;
    float4 v = *(const float4*)&h[(size_t)src[e] * Dc + q];
    float* ap = &agg[(size_t)dst[e] * Dc + q];
    atomicAdd(ap + 0, v.x);
    atomicAdd(ap + 1, v.y);
    atomicAdd(ap + 2, v.z);
    atomicAdd(ap + 3, v.w);
}

// pools[batch[n], col_off + d] += h[n, d]  (float4 per thread)
__global__ void k_pool(const float* __restrict__ h, const int* __restrict__ batch,
                       float* __restrict__ pools, int Nn, int Dc, int col_off, int LD) {
    int qn = Dc >> 2;
    int i = blockIdx.x * blockDim.x + threadIdx.x;
    if (i >= Nn * qn) return;
    int n = i / qn, q = (i - n * qn) << 2;
    float4 v = *(const float4*)&h[(size_t)n * Dc + q];
    float* pp = &pools[(size_t)batch[n] * LD + col_off + q];
    atomicAdd(pp + 0, v.x);
    atomicAdd(pp + 1, v.y);
    atomicAdd(pp + 2, v.z);
    atomicAdd(pp + 3, v.w);
}

// coalesced BN partial sums: per-column partials via LDS float atomics (ds_add_f32),
// then merged into global accumulators. Dc must be a power of two (<= 512).
__global__ __launch_bounds__(256)
void k_bn_partial(const float* __restrict__ z, float* __restrict__ gsum,
                  float* __restrict__ gsq, size_t total, int Dc) {
    __shared__ float ssum[512], ssq[512];
    for (int c = threadIdx.x; c < Dc; c += 256) { ssum[c] = 0.f; ssq[c] = 0.f; }
    __syncthreads();
    int mask = Dc - 1;
    for (size_t i = (size_t)blockIdx.x * 256 + threadIdx.x; i < total;
         i += (size_t)gridDim.x * 256) {
        float v = z[i];
        int c = (int)(i & mask);
        atomicAdd(&ssum[c], v);
        atomicAdd(&ssq[c], v * v);
    }
    __syncthreads();
    for (int c = threadIdx.x; c < Dc; c += 256) {
        atomicAdd(&gsum[c], ssum[c]);
        atomicAdd(&gsq[c], ssq[c]);
    }
}

__global__ void k_bn_finalize(const float* gsum, const float* gsq,
                              float* mean, float* rstd, int M, int Dc) {
    int c = blockIdx.x * blockDim.x + threadIdx.x;
    if (c >= Dc) return;
    float mu = gsum[c] / (float)M;
    float var = gsq[c] / (float)M - mu * mu;
    mean[c] = mu;
    rstd[c] = rsqrtf(var + BN_EPS);
}

// Dc must be a power of two
__global__ void k_bn_apply(float* z, const float* mean, const float* rstd,
                           const float* gamma, const float* beta, size_t total, int Dc, int relu) {
    size_t i = (size_t)blockIdx.x * blockDim.x + threadIdx.x;
    if (i >= total) return;
    int c = (int)(i & (size_t)(Dc - 1));
    float v = (z[i] - mean[c]) * rstd[c] * gamma[c] + beta[c];
    if (relu) v = fmaxf(v, 0.f);
    z[i] = v;
}

// e laid out [T][B][EMB] so each timestep is a contiguous [B,EMB] matrix
__global__ void k_embed(const int* __restrict__ smi, const float* __restrict__ emb,
                        float* __restrict__ e, int Bb, int T, int Emb) {
    int i = blockIdx.x * blockDim.x + threadIdx.x;
    if (i >= Bb * T * Emb) return;
    int d = i % Emb;
    int bt = i / Emb;
    int b = bt / T, t = bt % T;
    e[((size_t)t * Bb + b) * Emb + d] = emb[(size_t)smi[bt] * Emb + d];
}

__global__ void k_lstm_update(const float* __restrict__ gates, float* __restrict__ h,
                              float* __restrict__ c, int Bb, int Hh) {
    int i = blockIdx.x * blockDim.x + threadIdx.x;
    if (i >= Bb * Hh) return;
    int b = i / Hh, j = i % Hh;
    const float* g = gates + (size_t)b * 4 * Hh;
    float ig = g[j], fg = g[Hh + j], gg = g[2 * Hh + j], og = g[3 * Hh + j];
    float cc = c[i];
    cc = sigm(fg) * cc + sigm(ig) * tanhf(gg);
    c[i] = cc;
    h[i] = sigm(og) * tanhf(cc);
}

// L2-normalize each row; one block per row
__global__ void k_l2norm(float* out, int Nc) {
    int row = blockIdx.x;
    __shared__ float ls[256];
    float s = 0.f;
    for (int c = threadIdx.x; c < Nc; c += blockDim.x) {
        float v = out[(size_t)row * Nc + c];
        s += v * v;
    }
    ls[threadIdx.x] = s;
    __syncthreads();
    for (int o = 128; o > 0; o >>= 1) {
        if ((int)threadIdx.x < o) ls[threadIdx.x] += ls[threadIdx.x + o];
        __syncthreads();
    }
    float nrm = fmaxf(sqrtf(ls[0]), 1e-12f);
    for (int c = threadIdx.x; c < Nc; c += blockDim.x)
        out[(size_t)row * Nc + c] /= nrm;
}

// ---------------- generic WMMA GEMM (split-K concatenated A) ----------------
// C[M,N] = op( [A1 | A2] (+Aadd) @ B ) ; A1: fp32 [M,K1], A2: fp32 [M,K2] (K2 may be 0);
// Bt: bf16 transposed [N][K1+K2]; K1,K2 % 32 == 0; N % 128 == 0.
// Block = 256 threads (8 waves). Block tile 64(M) x 128(N), K-step 32, double-buffered LDS.
// Each wave computes a 32x32 sub-tile = 4 WMMAs per K-step.
// B tile staged with GLOBAL_LOAD_ASYNC_TO_LDS_B128, ASYNCcnt-fenced.
template <bool ADD, bool BIAS, bool RELU>
__global__ __launch_bounds__(256)
void k_gemm_wmma(const float* __restrict__ A1, int K1,
                 const float* __restrict__ A2, int K2,
                 const float* __restrict__ Aadd,
                 const unsigned short* __restrict__ Bt, const float* __restrict__ bias,
                 float* __restrict__ C, int M, int N) {
    __shared__ __align__(16) unsigned short sA[2][64][40];
    __shared__ __align__(16) unsigned short sBt[2][128][40];

    const int K = K1 + K2;
    const int row0 = blockIdx.y * 64;
    const int col0 = blockIdx.x * 128;
    const int tid  = threadIdx.x;
    const int lane = tid & 31;
    const int wave = tid >> 5;
    const int wm = (wave & 1) << 5;    // 0 / 32
    const int wn = (wave >> 1) << 5;   // 0 / 32 / 64 / 96
    const int hi = lane >> 4;          // 0 / 1
    const int lo = lane & 15;

    v8f acc00 = {}, acc01 = {}, acc10 = {}, acc11 = {};

    auto stageA = [&](int buf, int kk) {
        const float* P;
        int SK, kb;
        if (kk < K1) { P = A1; SK = K1; kb = kk; }
        else         { P = A2; SK = K2; kb = kk - K1; }
        #pragma unroll
        for (int i = 0; i < 2; ++i) {
            int e = tid + i * 256;            // 512 chunks of 4 floats (64 x 32)
            int r = e >> 3, kc = (e & 7) << 2;
            int row = row0 + r;
            float4 v = make_float4(0.f, 0.f, 0.f, 0.f);
            if (row < M) {
                v = *(const float4*)&P[(size_t)row * SK + kb + kc];
                if (ADD) {
                    float4 w2 = *(const float4*)&Aadd[(size_t)row * SK + kb + kc];
                    v.x += w2.x; v.y += w2.y; v.z += w2.z; v.w += w2.w;
                }
            }
            union { unsigned short s[4]; uint2 d; } pk;
            pk.s[0] = f2bf(v.x); pk.s[1] = f2bf(v.y); pk.s[2] = f2bf(v.z); pk.s[3] = f2bf(v.w);
            *(uint2*)&sA[buf][r][kc] = pk.d;
        }
    };
    auto stageB = [&](int buf, int kk) {
        #pragma unroll
        for (int i = 0; i < 2; ++i) {
            int e = tid + i * 256;            // 512 chunks of 8 bf16 (128 x 32)
            int n = e >> 2, kc = (e & 3) << 3;
            const unsigned short* gp = &Bt[(size_t)(col0 + n) * K + kk + kc];
            unsigned int lp = (unsigned int)(uintptr_t)&sBt[buf][n][kc];
            unsigned long long ga = (unsigned long long)(uintptr_t)gp;
            asm volatile("global_load_async_to_lds_b128 %0, %1, off"
                         :: "v"(lp), "v"(ga) : "memory");
        }
    };

    const int nsteps = K >> 5;
    stageA(0, 0);
    stageB(0, 0);
    asm volatile("s_wait_asynccnt 0x0" ::: "memory");
    __syncthreads();

    for (int s = 0; s < nsteps; ++s) {
        const int cur = s & 1;
        const int nxt = cur ^ 1;
        if (s + 1 < nsteps) {
            stageA(nxt, (s + 1) << 5);
            stageB(nxt, (s + 1) << 5);
        }

        FragU a0, a1, b0, b1;
        {
            const uint4* p0 = (const uint4*)&sA[cur][wm + lo][hi << 4];
            a0.q[0] = p0[0]; a0.q[1] = p0[1];
            const uint4* p1 = (const uint4*)&sA[cur][wm + 16 + lo][hi << 4];
            a1.q[0] = p1[0]; a1.q[1] = p1[1];
            const uint4* p2 = (const uint4*)&sBt[cur][wn + lo][hi << 4];
            b0.q[0] = p2[0]; b0.q[1] = p2[1];
            const uint4* p3 = (const uint4*)&sBt[cur][wn + 16 + lo][hi << 4];
            b1.q[0] = p3[0]; b1.q[1] = p3[1];
        }
        acc00 = __builtin_amdgcn_wmma_f32_16x16x32_bf16(false, a0.v, false, b0.v, (short)0, acc00, false, false);
        acc01 = __builtin_amdgcn_wmma_f32_16x16x32_bf16(false, a0.v, false, b1.v, (short)0, acc01, false, false);
        acc10 = __builtin_amdgcn_wmma_f32_16x16x32_bf16(false, a1.v, false, b0.v, (short)0, acc10, false, false);
        acc11 = __builtin_amdgcn_wmma_f32_16x16x32_bf16(false, a1.v, false, b1.v, (short)0, acc11, false, false);

        if (s + 1 < nsteps)
            asm volatile("s_wait_asynccnt 0x0" ::: "memory");
        __syncthreads();
    }

    // ---- store epilogue: fast path for full 8-row columns (no per-element guards) ----
    v8f* accs[4] = { &acc00, &acc01, &acc10, &acc11 };
    #pragma unroll
    for (int tm = 0; tm < 2; ++tm) {
        #pragma unroll
        for (int tn = 0; tn < 2; ++tn) {
            v8f acc = *accs[tm * 2 + tn];
            const int col = col0 + wn + (tn << 4) + lo;
            const float bv = BIAS ? bias[col] : 0.f;
            const int rbase = row0 + wm + (tm << 4) + (hi << 3);
            float* p = C + (size_t)rbase * N + col;
            if (rbase + 8 <= M) {
                #pragma unroll
                for (int r = 0; r < 8; ++r) {
                    float v = acc[r] + bv;
                    if (RELU) v = fmaxf(v, 0.f);
                    *p = v;
                    p += N;
                }
            } else {
                #pragma unroll
                for (int r = 0; r < 8; ++r) {
                    if (rbase + r < M) {
                        float v = acc[r] + bv;
                        if (RELU) v = fmaxf(v, 0.f);
                        *p = v;
                    }
                    p += N;
                }
            }
        }
    }
}

// ---------------- host orchestration ----------------
static inline dim3 gemm_grid(int M, int N) { return dim3((N + 127) / 128, (M + 63) / 64); }
static inline unsigned int nb(size_t n, int t) { return (unsigned int)((n + t - 1) / t); }

extern "C" void kernel_launch(void* const* d_in, const int* in_sizes, int n_in,
                              void* d_out, int out_size, void* d_ws, size_t ws_size,
                              hipStream_t stream) {
    (void)in_sizes; (void)n_in; (void)out_size; (void)ws_size;

    // ----- inputs (setup_inputs dict order) -----
    const float* x          = (const float*)d_in[0];
    const int*   edge_index = (const int*)  d_in[1];
    const int*   batch      = (const int*)  d_in[2];
    const int*   smi        = (const int*)  d_in[3];
    const float* gin0_w1    = (const float*)d_in[4];
    const float* gin0_b1    = (const float*)d_in[5];
    const float* gin0_w2    = (const float*)d_in[6];
    const float* gin0_b2    = (const float*)d_in[7];
    const float* ginr_w1    = (const float*)d_in[8];
    const float* ginr_b1    = (const float*)d_in[9];
    const float* ginr_w2    = (const float*)d_in[10];
    const float* ginr_b2    = (const float*)d_in[11];
    const float* bn_gamma   = (const float*)d_in[12];
    const float* bn_beta    = (const float*)d_in[13];
    const float* emb        = (const float*)d_in[14];
    const float* w_ih       = (const float*)d_in[15];
    const float* w_hh       = (const float*)d_in[16];
    const float* b_ih       = (const float*)d_in[17];
    const float* b_hh       = (const float*)d_in[18];
    const float* g_w1       = (const float*)d_in[19];
    const float* g_b1       = (const float*)d_in[20];
    const float* g_bn_gamma = (const float*)d_in[21];
    const float* g_bn_beta  = (const float*)d_in[22];
    const float* g_w2       = (const float*)d_in[23];

    const int* e_src = edge_index;          // [E]
    const int* e_dst = edge_index + PE;     // [E]

    // ----- workspace carve-up -----
    char* w = (char*)d_ws;
    auto alloc = [&](size_t bytes) -> char* {
        char* p = w;
        w += (bytes + 255) & ~(size_t)255;
        return p;
    };
    float* bufA   = (float*)alloc((size_t)PN * PD * 4);
    float* bufB   = (float*)alloc((size_t)PN * PD * 4);
    float* ztmp   = (float*)alloc((size_t)PN * PD * 4);
    float* pools  = (float*)alloc((size_t)PB * PH * 4);       // enc1 [B, L*D]
    float* eT     = (float*)alloc((size_t)PT * PB * PEMB * 4);// [T][B][EMB]
    float* lstm_h = (float*)alloc((size_t)PB * PH * 4);       // enc2 at end
    float* lstm_c = (float*)alloc((size_t)PB * PH * 4);
    float* gates  = (float*)alloc((size_t)PB * PH4 * 4);
    float* z1     = (float*)alloc((size_t)PB * 4 * PD * 4);   // [B, 512]
    float* gsum   = (float*)alloc(512 * 4);
    float* gsq    = (float*)alloc(512 * 4);
    float* mean   = (float*)alloc(512 * 4);
    float* rstd   = (float*)alloc(512 * 4);
    float* biascomb = (float*)alloc(PH4 * 4);
    // all weights stored bf16 TRANSPOSED: Bt[N][K]
    unsigned short* bw1_0 = (unsigned short*)alloc((size_t)PD * F_IN * 2);        // [128][32]
    unsigned short* bw2_0 = (unsigned short*)alloc((size_t)PD * PD * 2);          // [128][128]
    unsigned short* bw1_r = (unsigned short*)alloc((size_t)(PL - 1) * PD * PD * 2);
    unsigned short* bw2_r = (unsigned short*)alloc((size_t)(PL - 1) * PD * PD * 2);
    unsigned short* bwc   = (unsigned short*)alloc((size_t)PH4 * PKC * 2);        // [4H][EMB|H] combined
    unsigned short* bgw1  = (unsigned short*)alloc((size_t)4 * PD * PH * 2);      // [512][640]
    unsigned short* bgw2  = (unsigned short*)alloc((size_t)6 * PD * 4 * PD * 2);  // [768][512]

    float* g1 = (float*)d_out;                     // [B, 768]
    float* g2 = (float*)d_out + (size_t)PB * 6 * PD;

    const int TB = 256;
    const int BN_BLOCKS = 480;

    // ----- weight conversions (fp32 -> bf16, into Bt[N][K] layout) -----
    k_conv_bf16<<<nb((size_t)F_IN * PD, TB), TB, 0, stream>>>(gin0_w1, bw1_0, F_IN, PD, 1, F_IN, 0);
    k_conv_bf16<<<nb((size_t)PD * PD, TB), TB, 0, stream>>>(gin0_w2, bw2_0, PD, PD, 1, PD, 0);
    for (int l = 0; l < PL - 1; ++l) {
        k_conv_bf16<<<nb((size_t)PD * PD, TB), TB, 0, stream>>>(ginr_w1 + (size_t)l * PD * PD,
                                                                bw1_r + (size_t)l * PD * PD, PD, PD, 1, PD, 0);
        k_conv_bf16<<<nb((size_t)PD * PD, TB), TB, 0, stream>>>(ginr_w2 + (size_t)l * PD * PD,
                                                                bw2_r + (size_t)l * PD * PD, PD, PD, 1, PD, 0);
    }
    // combined LSTM weight: rows n of [4H]: cols [0,64) = w_ih[n,:], cols [64,704) = w_hh[n,:]
    k_conv_bf16<<<nb((size_t)PH4 * PEMB, TB), TB, 0, stream>>>(w_ih, bwc, PH4, PEMB, 0, PKC, 0);
    k_conv_bf16<<<nb((size_t)PH4 * PH, TB), TB, 0, stream>>>(w_hh, bwc, PH4, PH, 0, PKC, PEMB);
    k_conv_bf16<<<nb((size_t)PH * 4 * PD, TB), TB, 0, stream>>>(g_w1, bgw1, PH, 4 * PD, 1, PH, 0);
    k_conv_bf16<<<nb((size_t)4 * PD * 6 * PD, TB), TB, 0, stream>>>(g_w2, bgw2, 4 * PD, 6 * PD, 1, 4 * PD, 0);
    k_addvec<<<nb(PH4, TB), TB, 0, stream>>>(biascomb, b_ih, b_hh, PH4);

    // BN helper (zero accumulators, partial, finalize, apply)
    auto run_bn = [&](float* z, int M, int Dc, const float* gamma, const float* beta, int relu) {
        k_zero<<<nb((size_t)Dc, TB), TB, 0, stream>>>(gsum, (size_t)Dc);
        k_zero<<<nb((size_t)Dc, TB), TB, 0, stream>>>(gsq, (size_t)Dc);
        k_bn_partial<<<BN_BLOCKS, 256, 0, stream>>>(z, gsum, gsq, (size_t)M * Dc, Dc);
        k_bn_finalize<<<nb((size_t)Dc, TB), TB, 0, stream>>>(gsum, gsq, mean, rstd, M, Dc);
        k_bn_apply<<<nb((size_t)M * Dc, TB), TB, 0, stream>>>(z, mean, rstd, gamma, beta,
                                                              (size_t)M * Dc, Dc, relu);
    };

    // ----- GIN encoder -----
    k_zero<<<nb((size_t)PB * PH, TB), TB, 0, stream>>>(pools, (size_t)PB * PH);

    // layer 0: input x [N,32]
    k_zero<<<nb((size_t)PN * F_IN, TB), TB, 0, stream>>>(bufB, (size_t)PN * F_IN);
    k_edge_agg<<<nb((size_t)PE * (F_IN / 4), TB), TB, 0, stream>>>(x, e_src, e_dst, bufB, PE, F_IN);
    k_gemm_wmma<true, true, true><<<gemm_grid(PN, PD), 256, 0, stream>>>(
        x, F_IN, x, 0, bufB, bw1_0, gin0_b1, ztmp, PN, PD);
    k_gemm_wmma<false, true, true><<<gemm_grid(PN, PD), 256, 0, stream>>>(
        ztmp, PD, ztmp, 0, nullptr, bw2_0, gin0_b2, bufA, PN, PD);
    run_bn(bufA, PN, PD, bn_gamma, bn_beta, 0);
    k_pool<<<nb((size_t)PN * (PD / 4), TB), TB, 0, stream>>>(bufA, batch, pools, PN, PD, 0, PH);

    // layers 1..4 (ping-pong bufA/bufB)
    float* hcur = bufA;
    float* hnext = bufB;
    for (int l = 1; l < PL; ++l) {
        const unsigned short* w1 = bw1_r + (size_t)(l - 1) * PD * PD;
        const unsigned short* w2 = bw2_r + (size_t)(l - 1) * PD * PD;
        const float* b1 = ginr_b1 + (size_t)(l - 1) * PD;
        const float* b2 = ginr_b2 + (size_t)(l - 1) * PD;
        k_zero<<<nb((size_t)PN * PD, TB), TB, 0, stream>>>(hnext, (size_t)PN * PD);
        k_edge_agg<<<nb((size_t)PE * (PD / 4), TB), TB, 0, stream>>>(hcur, e_src, e_dst, hnext, PE, PD);
        k_gemm_wmma<true, true, true><<<gemm_grid(PN, PD), 256, 0, stream>>>(
            hcur, PD, hcur, 0, hnext, w1, b1, ztmp, PN, PD);
        k_gemm_wmma<false, true, true><<<gemm_grid(PN, PD), 256, 0, stream>>>(
            ztmp, PD, ztmp, 0, nullptr, w2, b2, hnext, PN, PD);
        run_bn(hnext, PN, PD, bn_gamma + (size_t)l * PD, bn_beta + (size_t)l * PD, 0);
        k_pool<<<nb((size_t)PN * (PD / 4), TB), TB, 0, stream>>>(hnext, batch, pools, PN, PD, l * PD, PH);
        float* t = hcur; hcur = hnext; hnext = t;
    }
    // enc1 == pools

    // ----- SMILES encoder: embed + LSTM (fused [xt|h] @ [Wih;Whh]^T per step) -----
    k_embed<<<nb((size_t)PB * PT * PEMB, TB), TB, 0, stream>>>(smi, emb, eT, PB, PT, PEMB);
    k_zero<<<nb((size_t)PB * PH, TB), TB, 0, stream>>>(lstm_h, (size_t)PB * PH);
    k_zero<<<nb((size_t)PB * PH, TB), TB, 0, stream>>>(lstm_c, (size_t)PB * PH);
    for (int t = 0; t < PT; ++t) {
        const float* xt = eT + (size_t)t * PB * PEMB;   // [B, EMB]
        k_gemm_wmma<false, true, false><<<gemm_grid(PB, PH4), 256, 0, stream>>>(
            xt, PEMB, lstm_h, PH, nullptr, bwc, biascomb, gates, PB, PH4);
        k_lstm_update<<<nb((size_t)PB * PH, TB), TB, 0, stream>>>(gates, lstm_h, lstm_c, PB, PH);
    }
    // enc2 == lstm_h

    // ----- projection head g (applied to enc1 then enc2) -----
    const float* encs[2] = { pools, lstm_h };
    float* outs[2] = { g1, g2 };
    for (int s = 0; s < 2; ++s) {
        k_gemm_wmma<false, true, false><<<gemm_grid(PB, 4 * PD), 256, 0, stream>>>(
            encs[s], PH, encs[s], 0, nullptr, bgw1, g_b1, z1, PB, 4 * PD);
        run_bn(z1, PB, 4 * PD, g_bn_gamma, g_bn_beta, 1);
        k_gemm_wmma<false, false, false><<<gemm_grid(PB, 6 * PD), 256, 0, stream>>>(
            z1, 4 * PD, z1, 0, nullptr, bgw2, nullptr, outs[s], PB, 6 * PD);
        k_l2norm<<<PB, 256, 0, stream>>>(outs[s], 6 * PD);
    }
}
